// DisentangledMultiHeadAttention_12068858102186
// MI455X (gfx1250) — matrix operand order repaired and checked by reading, not compiled, not run on hardware
//
#include <hip/hip_runtime.h>

typedef __attribute__((ext_vector_type(16))) __bf16 v16bf;
typedef __attribute__((ext_vector_type(8)))  __bf16 v8bf;
typedef __attribute__((ext_vector_type(8)))  float  v8f;

#define WMMA_BF16(a, b, c) \
  __builtin_amdgcn_wmma_f32_16x16x32_bf16(false, (a), false, (b), (short)0, (c), false, false)

static __device__ __forceinline__ __bf16 f2bf(float f) {
  union { float f; unsigned u; } in; in.f = f;
  unsigned u = in.u;
  unsigned r = u + 0x7FFFu + ((u >> 16) & 1u);   // round-to-nearest-even
  union { unsigned short s; __bf16 b; } out; out.s = (unsigned short)(r >> 16);
  return out.b;
}

static __device__ __forceinline__ v16bf cat8(v8bf lo, v8bf hi) {
  v16bf r;
#pragma unroll
  for (int i = 0; i < 8; ++i) { r[i] = lo[i]; r[i + 8] = hi[i]; }
  return r;
}

// LDS byte offset of a pointer into __shared__ memory.
// (flat LDS aperture keeps the wave-relative LDS offset in addr[31:0])
template <typename T>
static __device__ __forceinline__ unsigned lds_off(T* p) {
  return (unsigned)(unsigned long long)p;
}

// CDNA5 async DMA copy: 16 bytes global -> LDS per lane, tracked by ASYNCcnt.
static __device__ __forceinline__ void async_cp16(unsigned lds_addr, const void* g) {
  asm volatile("global_load_async_to_lds_b128 %0, %1, off"
               :: "v"(lds_addr), "v"((unsigned long long)g)
               : "memory");
}

// ---------------------------------------------------------------- f32 -> bf16
__global__ void cvt_f32_bf16(const float* __restrict__ in, __bf16* __restrict__ out, int n) {
  int i = blockIdx.x * blockDim.x + threadIdx.x;
  if (i < n) out[i] = f2bf(in[i]);
}

// ------------------------------------------------- C = A(MxK) @ W(NxK)^T + b
// A,W bf16 row-major; C f32 row-major. Tile: 128 rows x 128 cols per block,
// 8 waves; each wave owns a 16-row strip and all 128 cols (8 16x16 accums).
// K-step 32, double-buffered LDS staging via global_load_async_to_lds_b128:
//   per thread per k-step: 2 x 16B of A tile (128x32) + 2 x 16B of W tile (128x32).
__global__ __launch_bounds__(256) void gemm_bias_bf16(
    const __bf16* __restrict__ A, const __bf16* __restrict__ W,
    const float* __restrict__ bias, float* __restrict__ C,
    int M, int N, int K) {
  __shared__ __align__(16) __bf16 Asm[2][128 * 32];   // 2 x 8 KB
  __shared__ __align__(16) __bf16 Wsm[2][128 * 32];   // 2 x 8 KB

  const int tid  = threadIdx.x;
  const int lane = tid & 31;
  const int wave = tid >> 5;
  const int l16  = lane & 15;
  const int hi   = lane >> 4;                         // 0 or 1
  const int m0   = blockIdx.x * 128;
  const int n0   = blockIdx.y * 128;

  // copy chunks: tile chunk c of [128 x 32]: row = c>>2, col = (c&3)*8,
  // LDS byte off = c*16. Each thread copies chunks tid and tid+256 of A and W.
  const __bf16* gA0 = A + (size_t)(m0 + (tid >> 2)) * K + (tid & 3) * 8;
  const __bf16* gA1 = A + (size_t)(m0 + ((tid + 256) >> 2)) * K + (tid & 3) * 8;
  const __bf16* gW0 = W + (size_t)(n0 + (tid >> 2)) * K + (tid & 3) * 8;
  const __bf16* gW1 = W + (size_t)(n0 + ((tid + 256) >> 2)) * K + (tid & 3) * 8;

  const unsigned aBase0 = lds_off(&Asm[0][0]), aBase1 = lds_off(&Asm[1][0]);
  const unsigned wBase0 = lds_off(&Wsm[0][0]), wBase1 = lds_off(&Wsm[1][0]);
  const unsigned off0 = (unsigned)tid * 16u;
  const unsigned off1 = (unsigned)(tid + 256) * 16u;

  v8f acc[8];
#pragma unroll
  for (int t = 0; t < 8; ++t) acc[t] = (v8f){};

  // prime buffer 0
  async_cp16(aBase0 + off0, gA0);
  async_cp16(aBase0 + off1, gA1);
  async_cp16(wBase0 + off0, gW0);
  async_cp16(wBase0 + off1, gW1);

  const int nk = K >> 5;
  for (int i = 0; i < nk; ++i) {
    const int cur = i & 1;
    if (i + 1 < nk) {
      const int k1 = (i + 1) * 32;
      const unsigned ab = (cur ? aBase0 : aBase1);
      const unsigned wb = (cur ? wBase0 : wBase1);
      async_cp16(ab + off0, gA0 + k1);
      async_cp16(ab + off1, gA1 + k1);
      async_cp16(wb + off0, gW0 + k1);
      async_cp16(wb + off1, gW1 + k1);
      asm volatile("s_wait_asynccnt 0x4" ::: "memory");   // current tile done
    } else {
      asm volatile("s_wait_asynccnt 0x0" ::: "memory");
    }
    __syncthreads();

    const __bf16* As = &Asm[cur][0];
    const __bf16* Ws = &Wsm[cur][0];
    const __bf16* ar = As + (wave * 16 + l16) * 32 + hi * 16;
    v16bf af = cat8(*(const v8bf*)ar, *(const v8bf*)(ar + 8));
    const __bf16* w0 = Ws + l16 * 32 + hi * 16;
#pragma unroll
    for (int t = 0; t < 8; ++t) {
      v16bf bf_ = *(const v16bf*)(w0 + t * 16 * 32);
      acc[t] = WMMA_BF16(af, bf_, acc[t]);
    }
    __syncthreads();                                      // tile free for reuse
  }

#pragma unroll
  for (int t = 0; t < 8; ++t) {
    const int col = n0 + t * 16 + l16;
    const float bv = bias[col];
#pragma unroll
    for (int r = 0; r < 8; ++r) {
      int m = m0 + wave * 16 + r + hi * 8;
      C[(size_t)m * N + col] = acc[t][r] + bv;
    }
  }
}

// ------------------------------------------- split cqk/pk into q, kk=k+pk, kT
// q,kk: (B,H,T,hd) row-major bf16 ; kT: (B,H,hd,T) bf16
__global__ void fuse_qkk(const float* __restrict__ cqk, const float* __restrict__ pk,
                         __bf16* __restrict__ Qb, __bf16* __restrict__ KKb,
                         __bf16* __restrict__ KTb) {
  int i = blockIdx.x * blockDim.x + threadIdx.x;     // over B*H*T*hd = 4M
  if (i >= 4 * 16 * 1024 * 64) return;
  int d = i & 63;
  int t = (i >> 6) & 1023;
  int h = (i >> 16) & 15;
  int b = i >> 20;
  size_t rowc = (size_t)(b * 1024 + t) * 2048 + h * 64 + d;
  float qf = cqk[rowc];
  float kf = cqk[rowc + 1024];
  float pf = pk[(size_t)(b * 1024 + t) * 1024 + h * 64 + d];
  Qb[i]  = f2bf(qf);
  KKb[i] = f2bf(kf + pf);
  KTb[((size_t)(b * 16 + h) * 64 + d) * 1024 + t] = f2bf(kf);
}

// --------------------------------------------------------- attention per block
// One block = one (b,h) pair and one 16-row q strip.
// S = scale * q @ kk^T  (16 x 1024, K=64)  -> LDS, softmax, attn out (f32)
// O = P @ k             (16 x 64,  K=1024) via kT, K split over 8 waves.
#define T_SEQ 1024
#define HD 64
__global__ __launch_bounds__(256) void attn_kernel(
    const __bf16* __restrict__ Q, const __bf16* __restrict__ KK,
    const __bf16* __restrict__ KT, float* __restrict__ attn_out,
    __bf16* __restrict__ ctx, float scale) {
  __shared__ __align__(16) float  S[16][T_SEQ];   // 64 KB
  __shared__ __align__(16) __bf16 P[16][T_SEQ];   // 32 KB
  __shared__ __align__(16) float  O[16][HD];      //  4 KB

  const int lane = threadIdx.x & 31;
  const int wave = threadIdx.x >> 5;
  const int l16  = lane & 15;
  const int hi   = lane >> 4;

  const int qb = blockIdx.x & 63;                 // q strip within T
  const int bh = blockIdx.x >> 6;                 // (b*H + h)

  const __bf16* qmat  = Q  + (size_t)bh * T_SEQ * HD;
  const __bf16* kkmat = KK + (size_t)bh * T_SEQ * HD;
  const __bf16* ktmat = KT + (size_t)bh * HD * T_SEQ;

  // ---- stage 1: score strip
  const int mrow = qb * 16 + l16;
  const __bf16* arow = qmat + (size_t)mrow * HD + hi * 16;
  v16bf aq0 = cat8(*(const v8bf*)(arow),      *(const v8bf*)(arow + 8));
  v16bf aq1 = cat8(*(const v8bf*)(arow + 32), *(const v8bf*)(arow + 40));

#pragma unroll
  for (int j = 0; j < 8; ++j) {
    int ct = wave * 8 + j;                        // 16-wide column tile
    int key = ct * 16 + l16;
    const __bf16* kb = kkmat + (size_t)key * HD + hi * 16;
    v16bf b0 = *(const v16bf*)(kb);
    v16bf b1 = *(const v16bf*)(kb + 32);
    v8f acc = {};
    acc = WMMA_BF16(aq0, b0, acc);
    acc = WMMA_BF16(aq1, b1, acc);
#pragma unroll
    for (int r = 0; r < 8; ++r)
      S[r + hi * 8][ct * 16 + l16] = acc[r] * scale;
  }
  __syncthreads();

  // ---- stage 2: softmax over each row; write attn (f32) + P (bf16)
#pragma unroll
  for (int rr = 0; rr < 2; ++rr) {
    int r = wave * 2 + rr;
    float mx = -3.0e38f;
    for (int c = lane; c < T_SEQ; c += 32) mx = fmaxf(mx, S[r][c]);
#pragma unroll
    for (int m = 16; m >= 1; m >>= 1) mx = fmaxf(mx, __shfl_xor(mx, m, 32));
    float sum = 0.f;
    for (int c = lane; c < T_SEQ; c += 32) {
      float e = __expf(S[r][c] - mx);
      S[r][c] = e;
      sum += e;
    }
#pragma unroll
    for (int m = 16; m >= 1; m >>= 1) sum += __shfl_xor(sum, m, 32);
    float inv = 1.0f / sum;
    float* aout = attn_out + ((size_t)bh * T_SEQ + qb * 16 + r) * T_SEQ;
    for (int c = lane; c < T_SEQ; c += 32) {
      float p = S[r][c] * inv;
      aout[c] = p;
      P[r][c] = f2bf(p);
    }
  }
  for (int i = threadIdx.x; i < 16 * HD; i += 256) ((float*)O)[i] = 0.f;
  __syncthreads();

  // ---- stage 3: O = P @ k (via kT), K split across waves
  v8f o0 = {}, o1 = {}, o2 = {}, o3 = {};
#pragma unroll
  for (int s = 0; s < 4; ++s) {
    int k0 = (wave * 4 + s) * 32;
    const __bf16* prow = &P[l16][k0 + hi * 16];
    v16bf ap = cat8(*(const v8bf*)(prow), *(const v8bf*)(prow + 8));
    const __bf16* ktb = ktmat + k0 + hi * 16;
    v16bf b0 = *(const v16bf*)(ktb + (size_t)( 0 + l16) * T_SEQ);
    v16bf b1 = *(const v16bf*)(ktb + (size_t)(16 + l16) * T_SEQ);
    v16bf b2 = *(const v16bf*)(ktb + (size_t)(32 + l16) * T_SEQ);
    v16bf b3 = *(const v16bf*)(ktb + (size_t)(48 + l16) * T_SEQ);
    o0 = WMMA_BF16(ap, b0, o0);
    o1 = WMMA_BF16(ap, b1, o1);
    o2 = WMMA_BF16(ap, b2, o2);
    o3 = WMMA_BF16(ap, b3, o3);
  }
#pragma unroll
  for (int r = 0; r < 8; ++r) {
    int m = r + hi * 8;
    atomicAdd(&O[m][ 0 + l16], o0[r]);
    atomicAdd(&O[m][16 + l16], o1[r]);
    atomicAdd(&O[m][32 + l16], o2[r]);
    atomicAdd(&O[m][48 + l16], o3[r]);
  }
  __syncthreads();

  // ---- write ctx (B,T,H*hd) bf16
  int b = bh >> 4, h = bh & 15;
  for (int i = threadIdx.x; i < 16 * HD; i += 256) {
    int m = i >> 6, d = i & 63;
    ctx[(size_t)(b * T_SEQ + qb * 16 + m) * 1024 + h * 64 + d] = f2bf(O[m][d]);
  }
}

// ----------------------------------------------------------------------------
extern "C" void kernel_launch(void* const* d_in, const int* in_sizes, int n_in,
                              void* d_out, int out_size, void* d_ws, size_t ws_size,
                              hipStream_t stream) {
  const float* x  = (const float*)d_in[0];
  const float* pe = (const float*)d_in[1];
  const float* Wc = (const float*)d_in[2];
  const float* bc = (const float*)d_in[3];
  const float* Wp = (const float*)d_in[4];
  const float* bp = (const float*)d_in[5];
  const float* Wo = (const float*)d_in[6];
  const float* bo = (const float*)d_in[7];
  float* out = (float*)d_out;

  const int B = 4, T = 1024, D = 1024, H = 16;
  const int BT = B * T;                 // 4096
  const int nBTD = BT * D;              // 4194304

  size_t off = 0;
  char* ws = (char*)d_ws;
  auto alloc = [&](size_t bytes) -> char* {
    char* p = ws + off;
    off += (bytes + 255) & ~(size_t)255;
    return p;
  };
  __bf16* xb   = (__bf16*)alloc((size_t)nBTD * 2);
  __bf16* peb  = (__bf16*)alloc((size_t)nBTD * 2);
  __bf16* Wcb  = (__bf16*)alloc((size_t)2 * D * D * 2);
  __bf16* Wpb  = (__bf16*)alloc((size_t)D * D * 2);
  __bf16* Wob  = (__bf16*)alloc((size_t)D * D * 2);
  float*  cqk  = (float*)alloc((size_t)BT * 2 * D * 4);
  float*  pkf  = (float*)alloc((size_t)BT * D * 4);
  __bf16* Qb   = (__bf16*)alloc((size_t)nBTD * 2);
  __bf16* KKb  = (__bf16*)alloc((size_t)nBTD * 2);
  __bf16* KTb  = (__bf16*)alloc((size_t)nBTD * 2);
  __bf16* ctxb = (__bf16*)alloc((size_t)nBTD * 2);

  // 1. convert inputs to bf16
  cvt_f32_bf16<<<(nBTD + 255) / 256, 256, 0, stream>>>(x, xb, nBTD);
  cvt_f32_bf16<<<(nBTD + 255) / 256, 256, 0, stream>>>(pe, peb, nBTD);
  cvt_f32_bf16<<<(2 * D * D + 255) / 256, 256, 0, stream>>>(Wc, Wcb, 2 * D * D);
  cvt_f32_bf16<<<(D * D + 255) / 256, 256, 0, stream>>>(Wp, Wpb, D * D);
  cvt_f32_bf16<<<(D * D + 255) / 256, 256, 0, stream>>>(Wo, Wob, D * D);

  // 2. cqk = x @ Wc^T + bc   (4096 x 2048 x 1024)
  gemm_bias_bf16<<<dim3(BT / 128, (2 * D) / 128), 256, 0, stream>>>(
      xb, Wcb, bc, cqk, BT, 2 * D, D);
  // 3. pk = pe @ Wp^T + bp   (4096 x 1024 x 1024)
  gemm_bias_bf16<<<dim3(BT / 128, D / 128), 256, 0, stream>>>(
      peb, Wpb, bp, pkf, BT, D, D);

  // 4. build q, kk=k+pk, kT
  fuse_qkk<<<(nBTD + 255) / 256, 256, 0, stream>>>(cqk, pkf, Qb, KKb, KTb);

  // 5. attention: attn -> d_out[nBTD ...], ctx -> workspace
  attn_kernel<<<B * H * (T / 16), 256, 0, stream>>>(
      Qb, KKb, KTb, out + (size_t)nBTD, ctxb, 0.125f);

  // 6. final projection: out = ctx @ Wo^T + bo
  gemm_bias_bf16<<<dim3(BT / 128, D / 128), 256, 0, stream>>>(
      ctxb, Wob, bo, out, BT, D, D);
}
